// GCN_32624571580953
// MI455X (gfx1250) — matrix operand order repaired and checked by reading, not compiled, not run on hardware
//
#include <hip/hip_runtime.h>
#include <hip/hip_bf16.h>

#define N_NODES 50000
#define N_PAD   50016            // padded to multiple of 32 for the GEMM
#define N_EDGES 800000
#define HID 512
#define N_LAYERS 3

typedef __attribute__((ext_vector_type(16))) __bf16 v16bf;
typedef __attribute__((ext_vector_type(8)))  __bf16 v8bf;
typedef __attribute__((ext_vector_type(4)))  __bf16 v4bf;
typedef __attribute__((ext_vector_type(8)))  float  v8f;

// ---------------------------------------------------------------------------
// Degree / normalization
// ---------------------------------------------------------------------------
__global__ void k_deg_init(float* __restrict__ deg) {
    int i = blockIdx.x * blockDim.x + threadIdx.x;
    if (i < N_NODES) deg[i] = 1.0f;   // self-loop contributes 1
}

__global__ void k_deg_edges(const long long* __restrict__ ei, float* __restrict__ deg) {
    int e = blockIdx.x * blockDim.x + threadIdx.x;
    if (e < N_EDGES) {
        int d = (int)ei[(size_t)N_EDGES + e];
        atomicAdd(&deg[d], 1.0f);
    }
}

__global__ void k_rsqrt(float* __restrict__ deg) {
    int i = blockIdx.x * blockDim.x + threadIdx.x;
    if (i < N_NODES) deg[i] = rsqrtf(deg[i]);
}

// ---------------------------------------------------------------------------
// Transpose + convert weights:  Wt[l][n][k] = (bf16) W[l][k][n]
// ---------------------------------------------------------------------------
__global__ void k_wt(const float* __restrict__ W, __bf16* __restrict__ Wt) {
    int idx = blockIdx.x * blockDim.x + threadIdx.x;
    if (idx >= N_LAYERS * HID * HID) return;
    int l = idx / (HID * HID);
    int r = idx % (HID * HID);
    int n = r / HID;
    int k = r % HID;
    Wt[idx] = (__bf16)W[(size_t)l * HID * HID + (size_t)k * HID + n];
}

// ---------------------------------------------------------------------------
// x (fp32) -> bf16 activation buffer
// ---------------------------------------------------------------------------
__global__ __launch_bounds__(256)
void k_x2bf(const float* __restrict__ x, __bf16* __restrict__ xb) {
    int idx = blockIdx.x * blockDim.x + threadIdx.x;   // over N*HID/4
    if (idx >= N_NODES * HID / 4) return;
    float4 f = ((const float4*)x)[idx];
    v4bf b; b[0] = (__bf16)f.x; b[1] = (__bf16)f.y;
            b[2] = (__bf16)f.z; b[3] = (__bf16)f.w;
    *(v4bf*)(xb + (size_t)idx * 4) = b;
}

// ---------------------------------------------------------------------------
// GEMM: C[Np x 512] = A[Np x 512] (bf16) * W (bf16, pre-transposed Wt[n][k])
// Block: 256 threads = 8 waves; block tile = 32 rows x 512 cols.
// A staged to LDS via GLOBAL_LOAD_ASYNC_TO_LDS_B128 (ASYNCcnt path).
// Each wave: 2 row-tiles x 4 col-tiles = 8 accumulators; each B fragment
// feeds two v_wmma_f32_16x16x32_bf16 -> 8 WMMA per K-step per wave.
// ---------------------------------------------------------------------------
__global__ __launch_bounds__(256)
void k_gemm(const __bf16* __restrict__ A, const __bf16* __restrict__ Wt,
            float* __restrict__ C) {
    __shared__ __bf16 ldsA[32 * HID];   // 32 KB

    const int t    = threadIdx.x;
    const int row0 = blockIdx.x * 32;

    // Async-copy 32 rows x 1KB of bf16 A into LDS.
    // NOTE: derive the LDS byte address from &ldsA so the array address
    // escapes (ptrtoint) -- the memory-clobbering asm is then assumed to
    // write ldsA, keeping the ds_load fragment reads below alive.
    {
        const unsigned lds_base = (unsigned)(size_t)(void*)ldsA;  // addr[31:0]
        const __bf16*  gsrc     = A + (size_t)row0 * HID;
        #pragma unroll
        for (int i = t; i < 32 * HID / 8; i += 256) {      // 2048 x 16B chunks
            unsigned ldsoff = lds_base + (unsigned)(i * 16);
            const __bf16* g = gsrc + (size_t)i * 8;
            asm volatile("global_load_async_to_lds_b128 %0, %1, off"
                         :: "v"(ldsoff), "v"(g) : "memory");
        }
        asm volatile("s_wait_asynccnt 0" ::: "memory");
    }
    __syncthreads();

    const int wave  = t >> 5;      // 0..7
    const int lane  = t & 31;
    const int half  = lane >> 4;   // lane group 0/1
    const int l15   = lane & 15;
    const int nbase = wave * 64;   // 4 column tiles of 16

    v8f a00 = {}, a01 = {}, a02 = {}, a03 = {};
    v8f a10 = {}, a11 = {}, a12 = {}, a13 = {};

    const __bf16* arow0 = &ldsA[l15 * HID];          // row tile 0: m = lane&15
    const __bf16* arow1 = &ldsA[(16 + l15) * HID];   // row tile 1
    for (int k0 = 0; k0 < HID; k0 += 32) {
        // A fragments (K contiguous per lane half, k = 16*half + s): two 16B LDS loads each.
        union { v16bf v; v8bf h[2]; } fa0, fa1;
        const int klo = k0 + half * 16;
        fa0.h[0] = *(const v8bf*)(arow0 + klo);
        fa0.h[1] = *(const v8bf*)(arow0 + klo + 8);
        fa1.h[0] = *(const v8bf*)(arow1 + klo);
        fa1.h[1] = *(const v8bf*)(arow1 + klo + 8);

        // B fragments: lane = col n, K contiguous -> one 32B global load each.
        const int kb = k0 + half * 16;
        const __bf16* bcol = Wt + (size_t)(nbase + l15) * HID + kb;
        v16bf b0 = *(const v16bf*)(bcol);
        v16bf b1 = *(const v16bf*)(bcol + 16 * HID);
        v16bf b2 = *(const v16bf*)(bcol + 32 * HID);
        v16bf b3 = *(const v16bf*)(bcol + 48 * HID);

        a00 = __builtin_amdgcn_wmma_f32_16x16x32_bf16(false, fa0.v, false, b0, (short)0, a00, false, false);
        a10 = __builtin_amdgcn_wmma_f32_16x16x32_bf16(false, fa1.v, false, b0, (short)0, a10, false, false);
        a01 = __builtin_amdgcn_wmma_f32_16x16x32_bf16(false, fa0.v, false, b1, (short)0, a01, false, false);
        a11 = __builtin_amdgcn_wmma_f32_16x16x32_bf16(false, fa1.v, false, b1, (short)0, a11, false, false);
        a02 = __builtin_amdgcn_wmma_f32_16x16x32_bf16(false, fa0.v, false, b2, (short)0, a02, false, false);
        a12 = __builtin_amdgcn_wmma_f32_16x16x32_bf16(false, fa1.v, false, b2, (short)0, a12, false, false);
        a03 = __builtin_amdgcn_wmma_f32_16x16x32_bf16(false, fa0.v, false, b3, (short)0, a03, false, false);
        a13 = __builtin_amdgcn_wmma_f32_16x16x32_bf16(false, fa1.v, false, b3, (short)0, a13, false, false);
    }

    // C/D layout: lane l, vgpr i -> (m = i + 8*half, n = l&15)
    float* c0 = C + (size_t)row0 * HID + nbase + l15;
    float* c1 = C + (size_t)(row0 + 16) * HID + nbase + l15;
    #pragma unroll
    for (int i = 0; i < 8; ++i) {
        const size_t m = (size_t)(i + 8 * half) * HID;
        c0[m +  0] = a00[i];  c0[m + 16] = a01[i];
        c0[m + 32] = a02[i];  c0[m + 48] = a03[i];
        c1[m +  0] = a10[i];  c1[m + 16] = a11[i];
        c1[m + 32] = a12[i];  c1[m + 48] = a13[i];
    }
}

// ---------------------------------------------------------------------------
// agg = h * dinv^2  (self-loop term)
// ---------------------------------------------------------------------------
__global__ __launch_bounds__(256)
void k_self(const float* __restrict__ h, const float* __restrict__ dinv,
            float* __restrict__ agg) {
    int idx = blockIdx.x * blockDim.x + threadIdx.x;   // over N*HID/4
    if (idx >= N_NODES * HID / 4) return;
    int row = idx / (HID / 4);
    float s = dinv[row]; s *= s;
    float4 v = ((const float4*)h)[idx];
    v.x *= s; v.y *= s; v.z *= s; v.w *= s;
    ((float4*)agg)[idx] = v;
}

// ---------------------------------------------------------------------------
// Edge scatter: one wave32 per edge; agg[dst] += norm * h[src]
// h rows (102 MB fp32) are L2-resident; atomics resolve in L2.
// ---------------------------------------------------------------------------
__global__ __launch_bounds__(256)
void k_edges(const long long* __restrict__ ei, const float* __restrict__ dinv,
             const float* __restrict__ h, float* __restrict__ agg) {
    int w    = (blockIdx.x * blockDim.x + threadIdx.x) >> 5;   // edge id
    int lane = threadIdx.x & 31;
    if (w >= N_EDGES) return;
    int s = (int)ei[w];
    int d = (int)ei[(size_t)N_EDGES + w];
    float norm = dinv[s] * dinv[d];
    const float4* hs = (const float4*)(h + (size_t)s * HID);
    float* ad = agg + (size_t)d * HID;
    #pragma unroll
    for (int i = 0; i < 4; ++i) {
        int j = i * 32 + lane;          // float4 index 0..127
        float4 v = hs[j];
        atomicAdd(&ad[j * 4 + 0], v.x * norm);
        atomicAdd(&ad[j * 4 + 1], v.y * norm);
        atomicAdd(&ad[j * 4 + 2], v.z * norm);
        atomicAdd(&ad[j * 4 + 3], v.w * norm);
    }
}

// ---------------------------------------------------------------------------
// act (bf16) = relu(agg + bias)   -- next layer consumes bf16 directly
// ---------------------------------------------------------------------------
__global__ __launch_bounds__(256)
void k_act(const float* __restrict__ agg, const float* __restrict__ bias,
           __bf16* __restrict__ outb) {
    int idx = blockIdx.x * blockDim.x + threadIdx.x;   // over N*HID/4
    if (idx >= N_NODES * HID / 4) return;
    int col4 = (idx % (HID / 4)) * 4;
    float4 v = ((const float4*)agg)[idx];
    v.x += bias[col4 + 0]; v.y += bias[col4 + 1];
    v.z += bias[col4 + 2]; v.w += bias[col4 + 3];
    v4bf b;
    b[0] = (__bf16)(v.x > 0.f ? v.x : 0.f);
    b[1] = (__bf16)(v.y > 0.f ? v.y : 0.f);
    b[2] = (__bf16)(v.z > 0.f ? v.z : 0.f);
    b[3] = (__bf16)(v.w > 0.f ? v.w : 0.f);
    *(v4bf*)(outb + (size_t)idx * 4) = b;
}

// ---------------------------------------------------------------------------
// Final projection: out[i] = dot(act[i,:], W_out) + b_out   (wave per node)
// ---------------------------------------------------------------------------
__global__ __launch_bounds__(256)
void k_out(const __bf16* __restrict__ hb, const float* __restrict__ wout,
           const float* __restrict__ bout, float* __restrict__ out) {
    int w    = (blockIdx.x * blockDim.x + threadIdx.x) >> 5;   // node id
    int lane = threadIdx.x & 31;
    if (w >= N_NODES) return;
    const v8bf*   hr = (const v8bf*)(hb + (size_t)w * HID);    // 16 bf16 / lane
    const float4* wr = (const float4*)wout;
    float s = 0.f;
    #pragma unroll
    for (int i = 0; i < 2; ++i) {
        v8bf  a  = hr[lane * 2 + i];
        float4 w0 = wr[lane * 4 + i * 2 + 0];
        float4 w1 = wr[lane * 4 + i * 2 + 1];
        s += (float)a[0] * w0.x + (float)a[1] * w0.y +
             (float)a[2] * w0.z + (float)a[3] * w0.w +
             (float)a[4] * w1.x + (float)a[5] * w1.y +
             (float)a[6] * w1.z + (float)a[7] * w1.w;
    }
    #pragma unroll
    for (int off = 16; off; off >>= 1) s += __shfl_down(s, off, 32);
    if (lane == 0) out[w] = s + bout[0];
}

// ---------------------------------------------------------------------------
// Launcher
// ---------------------------------------------------------------------------
extern "C" void kernel_launch(void* const* d_in, const int* in_sizes, int n_in,
                              void* d_out, int out_size, void* d_ws, size_t ws_size,
                              hipStream_t stream) {
    const float*     x     = (const float*)d_in[0];
    const long long* ei    = (const long long*)d_in[1];   // int64 [2, E]
    const float*     Ws    = (const float*)d_in[2];       // [3,512,512]
    const float*     bs    = (const float*)d_in[3];       // [3,512]
    const float*     W_out = (const float*)d_in[4];       // [512,1]
    const float*     b_out = (const float*)d_in[5];       // [1]
    float*           out   = (float*)d_out;

    // Workspace layout (offsets in bytes)
    char* ws = (char*)d_ws;
    __bf16* Wt   = (__bf16*)ws;                                    // 1.5 MB
    float*  dinv = (float*)(ws + (size_t)2  * 1024 * 1024);        // 200 KB
    __bf16* abf  = (__bf16*)(ws + (size_t)4 * 1024 * 1024);        // 51.2 MB (padded)
    float*  tmp  = (float*)(ws + (size_t)60  * 1024 * 1024);       // 102.4 MB (padded)
    float*  agg  = (float*)(ws + (size_t)166 * 1024 * 1024);       // 102.4 MB

    const int nelem4 = N_NODES * HID / 4;

    // Degree + dinv (deg stored in dinv buffer, rsqrt'd in place)
    k_deg_init <<<(N_NODES + 255) / 256, 256, 0, stream>>>(dinv);
    k_deg_edges<<<(N_EDGES + 255) / 256, 256, 0, stream>>>(ei, dinv);
    k_rsqrt    <<<(N_NODES + 255) / 256, 256, 0, stream>>>(dinv);

    // Pre-convert weights (transposed bf16) and x (bf16)
    k_wt  <<<(N_LAYERS * HID * HID + 255) / 256, 256, 0, stream>>>(Ws, Wt);
    k_x2bf<<<(nelem4 + 255) / 256, 256, 0, stream>>>(x, abf);

    for (int l = 0; l < N_LAYERS; ++l) {
        k_gemm <<<N_PAD / 32, 256, 0, stream>>>(abf, Wt + (size_t)l * HID * HID, tmp);
        k_self <<<(nelem4 + 255) / 256, 256, 0, stream>>>(tmp, dinv, agg);
        k_edges<<<(N_EDGES * 32) / 256, 256, 0, stream>>>(ei, dinv, tmp, agg);
        k_act  <<<(nelem4 + 255) / 256, 256, 0, stream>>>(agg, bs + (size_t)l * HID, abf);
    }

    k_out<<<(N_NODES * 32 + 255) / 256, 256, 0, stream>>>(abf, W_out, b_out, out);
}
